// FusionDTI_41515153883347
// MI455X (gfx1250) — compile-verified, matrix operand order checked by reading, not run on hardware
//
#include <hip/hip_runtime.h>
#include <hip/hip_bf16.h>

// ---------------------------------------------------------------------------
// FusionDTI forward for MI455X (gfx1250, wave32, WMMA).
//
// Roofline reasoning:
//  * group-mean commutes with the linear projection -> project pooled tokens
//    (8x fewer GEMM FLOPs; embedding read ~386MB ~= 17us @ 23.3 TB/s is the
//    bandwidth floor).
//  * all GEMMs + attention run on v_wmma_f32_16x16x32_bf16 (bf16 in, f32 acc).
//  * masked softmax + masked-mean are fused into the attention kernel; the
//    pooled embedding is accumulated with global f32 atomics directly into the
//    concatenated MLP input.
// ---------------------------------------------------------------------------

typedef __attribute__((ext_vector_type(16))) __bf16          v16bf;
typedef __attribute__((ext_vector_type(8)))  float           v8f;
typedef __attribute__((ext_vector_type(16))) unsigned short  v16u;
typedef __attribute__((ext_vector_type(8)))  unsigned short  u16x8;
typedef __attribute__((ext_vector_type(4)))  unsigned short  u16x4;

#define B_    32
#define LP_   2048
#define LD_   512
#define DP_   1280
#define DD_   768
#define GP_   256
#define GD_   64
#define HID_  512
#define H_    8
#define HD_   64

__device__ __forceinline__ unsigned short f2bf(float f) {
    unsigned int u = __float_as_uint(f);
    unsigned int r = u + 0x7fffu + ((u >> 16) & 1u);   // round-to-nearest-even
    return (unsigned short)(r >> 16);
}

// Build a 16x32 bf16 fragment (A layout; B uses same packing on [N,K] data).
// p points at row_base + half*8 (elements).  element j <-> K = (j/8)*16 + j%8.
__device__ __forceinline__ v16bf frag_global(const unsigned short* p) {
    u16x8 lo = *(const u16x8*)(p);
    u16x8 hi = *(const u16x8*)(p + 16);
    v16u t;
#pragma unroll
    for (int i = 0; i < 8; ++i) { t[i] = lo[i]; t[i + 8] = hi[i]; }
    return __builtin_bit_cast(v16bf, t);
}

__device__ __forceinline__ v16bf frag_lds_f32(const float* rowp, int k0, int half) {
    v16u t;
#pragma unroll
    for (int j = 0; j < 16; ++j) {
        int k = k0 + ((j >> 3) << 4) + half * 8 + (j & 7);
        t[j] = f2bf(rowp[k]);
    }
    return __builtin_bit_cast(v16bf, t);
}

// --------------------------- prep kernels ----------------------------------

// mean over GROUP=8 rows, f32 -> bf16.  One thread per 4 output features.
__global__ void group_mean_kernel(const float* __restrict__ src,
                                  unsigned short* __restrict__ dst,
                                  int BG, int D4) {
    size_t idx = (size_t)blockIdx.x * blockDim.x + threadIdx.x;
    if (idx >= (size_t)BG * D4) return;
    int    d4 = (int)(idx % D4);
    size_t bg = idx / D4;
    const float4* s = (const float4*)src;
    size_t base = bg * 8 * (size_t)D4 + d4;
    float4 acc = make_float4(0.f, 0.f, 0.f, 0.f);
#pragma unroll
    for (int r = 0; r < 8; ++r) {
        float4 v = s[base + (size_t)r * D4];
        acc.x += v.x; acc.y += v.y; acc.z += v.z; acc.w += v.w;
    }
    u16x4 o;
    o[0] = f2bf(acc.x * 0.125f); o[1] = f2bf(acc.y * 0.125f);
    o[2] = f2bf(acc.z * 0.125f); o[3] = f2bf(acc.w * 0.125f);
    *(u16x4*)(dst + idx * 4) = o;
}

// group mask = any(8 bools); also count valid groups per batch (f32 atomics).
__global__ void group_mask_kernel(const unsigned char* __restrict__ m,
                                  float* __restrict__ mg,
                                  float* __restrict__ cnt, int G) {
    int idx = blockIdx.x * blockDim.x + threadIdx.x;
    if (idx >= B_ * G) return;
    const unsigned char* p = m + (size_t)idx * 8;
    int any = 0;
#pragma unroll
    for (int r = 0; r < 8; ++r) any |= p[r];
    mg[idx] = any ? 1.f : 0.f;
    if (any) atomicAdd(&cnt[idx / G], 1.f);
}

// w[K,N] f32 -> wt[N,K] bf16 (column-major B operand).
__global__ void transpose_bf16_kernel(const float* __restrict__ w,
                                      unsigned short* __restrict__ wt,
                                      int K, int N) {
    size_t idx = (size_t)blockIdx.x * blockDim.x + threadIdx.x;
    if (idx >= (size_t)K * N) return;
    int k = (int)(idx / N), n = (int)(idx % N);
    wt[(size_t)n * K + k] = f2bf(w[idx]);
}

__global__ void zero_kernel(float* __restrict__ p, int n) {
    int i = blockIdx.x * blockDim.x + threadIdx.x;
    if (i < n) p[i] = 0.f;
}

// ------------------------------ WMMA GEMM ----------------------------------
// C = A[M,K](bf16) * Bt[N,K](bf16)^T  (+bias), f32 accumulate, bf16 store.
// mode 0: out row-major [M,N]
// mode 1: out [B,H,L,64]   (Q/K head layout),  m = b*L + l, n = h*64+d
// mode 2: out [B,H,64,L]   (V transposed),     same decomposition
__global__ __launch_bounds__(256)
void gemm_bf16_wmma(const unsigned short* __restrict__ A,
                    const unsigned short* __restrict__ Bt,
                    const float* __restrict__ bias,
                    unsigned short* __restrict__ out,
                    int M, int N, int K, int mode, int Lrows) {
    int nt16  = N >> 4;
    int tiles = (M >> 4) * nt16;
    int tile  = blockIdx.x * 8 + (threadIdx.x >> 5);
    if (tile >= tiles) return;                       // uniform per wave
    int mT = tile / nt16, nT = tile % nt16;
    int lane = threadIdx.x & 31, row = lane & 15, half = lane >> 4;

    const unsigned short* Ap = A  + (size_t)(mT * 16 + row) * K + half * 8;
    const unsigned short* Bp = Bt + (size_t)(nT * 16 + row) * K + half * 8;

    v8f acc = {};
    for (int k0 = 0; k0 < K; k0 += 32) {
        __builtin_prefetch(Ap + k0 + 128, 0, 0);     // global_prefetch_b8
        __builtin_prefetch(Bp + k0 + 128, 0, 0);
        v16bf a = frag_global(Ap + k0);
        v16bf b = frag_global(Bp + k0);
        acc = __builtin_amdgcn_wmma_f32_16x16x32_bf16(
            false, a, false, b, (short)0, acc, false, false);
    }

    int col  = nT * 16 + row;
    float bv = bias ? bias[col] : 0.f;
#pragma unroll
    for (int v = 0; v < 8; ++v) {
        int   r   = mT * 16 + v + 8 * half;
        float val = acc[v] + bv;
        size_t oidx;
        if (mode == 0) {
            oidx = (size_t)r * N + col;
        } else {
            int bb = r / Lrows, l = r % Lrows, h = col >> 6, d = col & 63;
            if (mode == 1) oidx = ((((size_t)bb * H_ + h) * Lrows + l) << 6) + d;
            else           oidx = (((size_t)bb * H_ + h) * 64 + d) * (size_t)Lrows + l;
        }
        out[oidx] = f2bf(val);
    }
}

// --------------------------- fused attention -------------------------------
// One wave per (b, h, 16-query block).  Computes both pairwise softmaxes
// (protein keys L2a, drug keys L2b), the 0.5*(P1@V1 + P2@V2) mix, and the
// masked-mean pooling (atomicAdd into x[b, xoff + h*64 + d]).
__global__ __launch_bounds__(32)
void attn_wmma_kernel(const unsigned short* __restrict__ Q,   // [B,H,L1,64]
                      const unsigned short* __restrict__ K1,  // [B,H,L2a,64]
                      const unsigned short* __restrict__ V1t, // [B,H,64,L2a]
                      const float* __restrict__ mcol1, int L2a,
                      const unsigned short* __restrict__ K2,  // [B,H,L2b,64]
                      const unsigned short* __restrict__ V2t, // [B,H,64,L2b]
                      const float* __restrict__ mcol2, int L2b,
                      const float* __restrict__ mrow,
                      const float* __restrict__ cnt,
                      float* __restrict__ xout, int L1, int xoff) {
    __shared__ float P1[16][260];   // padded to avoid bank conflicts
    __shared__ float P2[16][68];

    int qblocks = L1 >> 4;
    int id = blockIdx.x;
    int qb = id % qblocks;
    int h  = (id / qblocks) % H_;
    int bb = id / (qblocks * H_);
    int lane = threadIdx.x, row = lane & 15, half = lane >> 4;

    // Q fragments for the two K-steps of the 64-wide head dim.
    const unsigned short* Qp =
        Q + ((((size_t)bb * H_ + h) * L1 + qb * 16 + row) << 6) + half * 8;
    v16bf aq0 = frag_global(Qp);
    v16bf aq1 = frag_global(Qp + 32);

    // ---- logits vs K1 (protein) keys ----
    for (int t = 0; t < (L2a >> 4); ++t) {
        const unsigned short* Kp =
            K1 + ((((size_t)bb * H_ + h) * L2a + t * 16 + row) << 6) + half * 8;
        v8f c = {};
        c = __builtin_amdgcn_wmma_f32_16x16x32_bf16(false, aq0, false,
                frag_global(Kp), (short)0, c, false, false);
        c = __builtin_amdgcn_wmma_f32_16x16x32_bf16(false, aq1, false,
                frag_global(Kp + 32), (short)0, c, false, false);
        int   l2 = t * 16 + row;
        float cm = mcol1[bb * L2a + l2];
#pragma unroll
        for (int v = 0; v < 8; ++v) {
            int   r  = v + 8 * half;
            float mr = mrow[bb * L1 + qb * 16 + r];
            P1[r][l2] = c[v] - ((cm > 0.f && mr > 0.f) ? 0.f : 1e6f);
        }
    }
    // ---- logits vs K2 (drug) keys ----
    for (int t = 0; t < (L2b >> 4); ++t) {
        const unsigned short* Kp =
            K2 + ((((size_t)bb * H_ + h) * L2b + t * 16 + row) << 6) + half * 8;
        v8f c = {};
        c = __builtin_amdgcn_wmma_f32_16x16x32_bf16(false, aq0, false,
                frag_global(Kp), (short)0, c, false, false);
        c = __builtin_amdgcn_wmma_f32_16x16x32_bf16(false, aq1, false,
                frag_global(Kp + 32), (short)0, c, false, false);
        int   l2 = t * 16 + row;
        float cm = mcol2[bb * L2b + l2];
#pragma unroll
        for (int v = 0; v < 8; ++v) {
            int   r  = v + 8 * half;
            float mr = mrow[bb * L1 + qb * 16 + r];
            P2[r][l2] = c[v] - ((cm > 0.f && mr > 0.f) ? 0.f : 1e6f);
        }
    }
    __syncthreads();

    // ---- row softmax: lanes 0-15 handle P1 rows, lanes 16-31 handle P2 ----
    if (lane < 16) {
        float* pr = &P1[lane][0];
        float mx = -1e30f;
        for (int i = 0; i < L2a; ++i) mx = fmaxf(mx, pr[i]);
        float s = 0.f;
        for (int i = 0; i < L2a; ++i) { float e = __expf(pr[i] - mx); pr[i] = e; s += e; }
        float inv = 1.f / s;
        for (int i = 0; i < L2a; ++i) pr[i] *= inv;
    } else {
        float* pr = &P2[lane - 16][0];
        float mx = -1e30f;
        for (int i = 0; i < L2b; ++i) mx = fmaxf(mx, pr[i]);
        float s = 0.f;
        for (int i = 0; i < L2b; ++i) { float e = __expf(pr[i] - mx); pr[i] = e; s += e; }
        float inv = 1.f / s;
        for (int i = 0; i < L2b; ++i) pr[i] *= inv;
    }
    __syncthreads();

    // ---- mix: 0.5*(P1@V1 + P2@V2), masked-mean pooled into xout ----
    float cb = cnt[bb];
#pragma unroll
    for (int nt = 0; nt < 4; ++nt) {
        v8f c = {};
        const unsigned short* Vp1 =
            V1t + ((((size_t)bb * H_ + h) * 64 + nt * 16 + row) * (size_t)L2a) + half * 8;
        for (int k0 = 0; k0 < L2a; k0 += 32) {
            v16bf ap = frag_lds_f32(&P1[row][0], k0, half);
            c = __builtin_amdgcn_wmma_f32_16x16x32_bf16(
                false, ap, false, frag_global(Vp1 + k0), (short)0, c, false, false);
        }
        const unsigned short* Vp2 =
            V2t + ((((size_t)bb * H_ + h) * 64 + nt * 16 + row) * (size_t)L2b) + half * 8;
        for (int k0 = 0; k0 < L2b; k0 += 32) {
            v16bf ap = frag_lds_f32(&P2[row][0], k0, half);
            c = __builtin_amdgcn_wmma_f32_16x16x32_bf16(
                false, ap, false, frag_global(Vp2 + k0), (short)0, c, false, false);
        }
        int d = nt * 16 + row;
#pragma unroll
        for (int v = 0; v < 8; ++v) {
            int   l1 = qb * 16 + v + 8 * half;
            float sc = mrow[bb * L1 + l1] * 0.5f / cb;
            atomicAdd(&xout[bb * 1024 + xoff + h * 64 + d], c[v] * sc);
        }
    }
}

// ------------------------------ MLP tail -----------------------------------

__global__ void fc_kernel(const float* __restrict__ X, const float* __restrict__ W,
                          const float* __restrict__ bias, float* __restrict__ Y,
                          int M, int K, int N, int relu) {
    int idx = blockIdx.x * blockDim.x + threadIdx.x;
    if (idx >= M * N) return;
    int m = idx / N, n = idx % N;
    float acc = bias[n];
    for (int k = 0; k < K; ++k) acc += X[(size_t)m * K + k] * W[(size_t)k * N + n];
    Y[idx] = relu ? fmaxf(acc, 0.f) : acc;
}

__global__ void bn_kernel(float* __restrict__ X, const float* __restrict__ g,
                          const float* __restrict__ be, int M, int N) {
    int n = blockIdx.x * blockDim.x + threadIdx.x;
    if (n >= N) return;
    float mu = 0.f;
    for (int m = 0; m < M; ++m) mu += X[(size_t)m * N + n];
    mu /= M;
    float var = 0.f;
    for (int m = 0; m < M; ++m) { float d = X[(size_t)m * N + n] - mu; var += d * d; }
    var /= M;
    float is = rsqrtf(var + 1e-5f);
    for (int m = 0; m < M; ++m)
        X[(size_t)m * N + n] = g[n] * (X[(size_t)m * N + n] - mu) * is + be[n];
}

__global__ void head_kernel(const float* __restrict__ X, const float* __restrict__ wo,
                            const float* __restrict__ bo, float* __restrict__ out,
                            int M, int K) {
    int m = blockIdx.x * blockDim.x + threadIdx.x;
    if (m >= M) return;
    float acc = bo[0];
    for (int k = 0; k < K; ++k) acc += X[(size_t)m * K + k] * wo[k];
    out[m] = 1.f / (1.f + __expf(-acc));
}

// ------------------------------- launch ------------------------------------

extern "C" void kernel_launch(void* const* d_in, const int* in_sizes, int n_in,
                              void* d_out, int out_size, void* d_ws, size_t ws_size,
                              hipStream_t stream) {
    const float*         prot   = (const float*)d_in[0];
    const float*         drug   = (const float*)d_in[1];
    const unsigned char* pmask  = (const unsigned char*)d_in[2];
    const unsigned char* dmask  = (const unsigned char*)d_in[3];
    const float* w_preg = (const float*)d_in[4];  const float* b_preg = (const float*)d_in[5];
    const float* w_dreg = (const float*)d_in[6];  const float* b_dreg = (const float*)d_in[7];
    const float* wqp = (const float*)d_in[8];  const float* wkp = (const float*)d_in[9];
    const float* wvp = (const float*)d_in[10]; const float* wqd = (const float*)d_in[11];
    const float* wkd = (const float*)d_in[12]; const float* wvd = (const float*)d_in[13];
    const float* w1 = (const float*)d_in[14]; const float* b1 = (const float*)d_in[15];
    const float* g1 = (const float*)d_in[16]; const float* be1 = (const float*)d_in[17];
    const float* w2 = (const float*)d_in[18]; const float* b2 = (const float*)d_in[19];
    const float* g2 = (const float*)d_in[20]; const float* be2 = (const float*)d_in[21];
    const float* w3 = (const float*)d_in[22]; const float* b3 = (const float*)d_in[23];
    const float* g3 = (const float*)d_in[24]; const float* be3 = (const float*)d_in[25];
    const float* wo = (const float*)d_in[26]; const float* bo = (const float*)d_in[27];

    char* W = (char*)d_ws;
    size_t off = 0;
    auto take = [&](size_t elems, size_t esz) {
        size_t o = off;
        off += ((elems * esz + 255) & ~(size_t)255);
        return o;
    };
    unsigned short* pgm  = (unsigned short*)(W + take((size_t)B_ * GP_ * DP_, 2));
    unsigned short* dgm  = (unsigned short*)(W + take((size_t)B_ * GD_ * DD_, 2));
    unsigned short* wpt  = (unsigned short*)(W + take((size_t)HID_ * DP_, 2));
    unsigned short* wdt  = (unsigned short*)(W + take((size_t)HID_ * DD_, 2));
    unsigned short* wqpt = (unsigned short*)(W + take((size_t)HID_ * HID_, 2));
    unsigned short* wkpt = (unsigned short*)(W + take((size_t)HID_ * HID_, 2));
    unsigned short* wvpt = (unsigned short*)(W + take((size_t)HID_ * HID_, 2));
    unsigned short* wqdt = (unsigned short*)(W + take((size_t)HID_ * HID_, 2));
    unsigned short* wkdt = (unsigned short*)(W + take((size_t)HID_ * HID_, 2));
    unsigned short* wvdt = (unsigned short*)(W + take((size_t)HID_ * HID_, 2));
    unsigned short* pg   = (unsigned short*)(W + take((size_t)B_ * GP_ * HID_, 2));
    unsigned short* dg   = (unsigned short*)(W + take((size_t)B_ * GD_ * HID_, 2));
    unsigned short* qp   = (unsigned short*)(W + take((size_t)B_ * GP_ * HID_, 2));
    unsigned short* kp   = (unsigned short*)(W + take((size_t)B_ * GP_ * HID_, 2));
    unsigned short* vpt  = (unsigned short*)(W + take((size_t)B_ * GP_ * HID_, 2));
    unsigned short* qd   = (unsigned short*)(W + take((size_t)B_ * GD_ * HID_, 2));
    unsigned short* kd   = (unsigned short*)(W + take((size_t)B_ * GD_ * HID_, 2));
    unsigned short* vdt  = (unsigned short*)(W + take((size_t)B_ * GD_ * HID_, 2));
    float* mp   = (float*)(W + take((size_t)B_ * GP_, 4));
    float* md   = (float*)(W + take((size_t)B_ * GD_, 4));
    float* cntp = (float*)(W + take(B_, 4));
    float* cntd = (float*)(W + take(B_, 4));
    float* x    = (float*)(W + take((size_t)B_ * 1024, 4));
    float* h1   = (float*)(W + take((size_t)B_ * 1024, 4));
    float* h2   = (float*)(W + take((size_t)B_ * 512, 4));
    float* h3   = (float*)(W + take((size_t)B_ * 256, 4));

    auto blk = [](size_t n, int t) { return dim3((unsigned)((n + t - 1) / t)); };

    // zero accumulators (x is atomically accumulated; counts too)
    zero_kernel<<<blk(B_ * 1024, 256), 256, 0, stream>>>(x, B_ * 1024);
    zero_kernel<<<1, 64, 0, stream>>>(cntp, B_);
    zero_kernel<<<1, 64, 0, stream>>>(cntd, B_);

    // group means (pooling before projection — 8x GEMM FLOP reduction)
    group_mean_kernel<<<blk((size_t)B_ * GP_ * (DP_ / 4), 256), 256, 0, stream>>>(
        prot, pgm, B_ * GP_, DP_ / 4);
    group_mean_kernel<<<blk((size_t)B_ * GD_ * (DD_ / 4), 256), 256, 0, stream>>>(
        drug, dgm, B_ * GD_, DD_ / 4);
    group_mask_kernel<<<blk(B_ * GP_, 256), 256, 0, stream>>>(pmask, mp, cntp, GP_);
    group_mask_kernel<<<blk(B_ * GD_, 256), 256, 0, stream>>>(dmask, md, cntd, GD_);

    // weights -> bf16 [N,K]
    transpose_bf16_kernel<<<blk((size_t)DP_ * HID_, 256), 256, 0, stream>>>(w_preg, wpt, DP_, HID_);
    transpose_bf16_kernel<<<blk((size_t)DD_ * HID_, 256), 256, 0, stream>>>(w_dreg, wdt, DD_, HID_);
    transpose_bf16_kernel<<<blk((size_t)HID_ * HID_, 256), 256, 0, stream>>>(wqp, wqpt, HID_, HID_);
    transpose_bf16_kernel<<<blk((size_t)HID_ * HID_, 256), 256, 0, stream>>>(wkp, wkpt, HID_, HID_);
    transpose_bf16_kernel<<<blk((size_t)HID_ * HID_, 256), 256, 0, stream>>>(wvp, wvpt, HID_, HID_);
    transpose_bf16_kernel<<<blk((size_t)HID_ * HID_, 256), 256, 0, stream>>>(wqd, wqdt, HID_, HID_);
    transpose_bf16_kernel<<<blk((size_t)HID_ * HID_, 256), 256, 0, stream>>>(wkd, wkdt, HID_, HID_);
    transpose_bf16_kernel<<<blk((size_t)HID_ * HID_, 256), 256, 0, stream>>>(wvd, wvdt, HID_, HID_);

    // regression projections on pooled tokens
    {
        int M = B_ * GP_, N = HID_, K = DP_;
        int tiles = (M / 16) * (N / 16);
        gemm_bf16_wmma<<<blk(tiles, 8), 256, 0, stream>>>(pgm, wpt, b_preg, pg, M, N, K, 0, GP_);
    }
    {
        int M = B_ * GD_, N = HID_, K = DD_;
        int tiles = (M / 16) * (N / 16);
        gemm_bf16_wmma<<<blk(tiles, 8), 256, 0, stream>>>(dgm, wdt, b_dreg, dg, M, N, K, 0, GD_);
    }

    // q/k/v projections
    {
        int M = B_ * GP_, N = HID_, K = HID_;
        int tiles = (M / 16) * (N / 16);
        gemm_bf16_wmma<<<blk(tiles, 8), 256, 0, stream>>>(pg, wqpt, nullptr, qp, M, N, K, 1, GP_);
        gemm_bf16_wmma<<<blk(tiles, 8), 256, 0, stream>>>(pg, wkpt, nullptr, kp, M, N, K, 1, GP_);
        gemm_bf16_wmma<<<blk(tiles, 8), 256, 0, stream>>>(pg, wvpt, nullptr, vpt, M, N, K, 2, GP_);
    }
    {
        int M = B_ * GD_, N = HID_, K = HID_;
        int tiles = (M / 16) * (N / 16);
        gemm_bf16_wmma<<<blk(tiles, 8), 256, 0, stream>>>(dg, wqdt, nullptr, qd, M, N, K, 1, GD_);
        gemm_bf16_wmma<<<blk(tiles, 8), 256, 0, stream>>>(dg, wkdt, nullptr, kd, M, N, K, 1, GD_);
        gemm_bf16_wmma<<<blk(tiles, 8), 256, 0, stream>>>(dg, wvdt, nullptr, vdt, M, N, K, 2, GD_);
    }

    // fused attention + masked-mean pooling into x[:, :512] and x[:, 512:]
    attn_wmma_kernel<<<dim3(B_ * H_ * (GP_ / 16)), 32, 0, stream>>>(
        qp, kp, vpt, mp, GP_, kd, vdt, md, GD_, mp, cntp, x, GP_, 0);
    attn_wmma_kernel<<<dim3(B_ * H_ * (GD_ / 16)), 32, 0, stream>>>(
        qd, kp, vpt, mp, GP_, kd, vdt, md, GD_, md, cntd, x, GD_, 512);

    // MLP tail (tiny: M=32)
    fc_kernel<<<blk(B_ * 1024, 256), 256, 0, stream>>>(x, w1, b1, h1, B_, 1024, 1024, 1);
    bn_kernel<<<blk(1024, 256), 256, 0, stream>>>(h1, g1, be1, B_, 1024);
    fc_kernel<<<blk(B_ * 512, 256), 256, 0, stream>>>(h1, w2, b2, h2, B_, 1024, 512, 1);
    bn_kernel<<<blk(512, 256), 256, 0, stream>>>(h2, g2, be2, B_, 512);
    fc_kernel<<<blk(B_ * 256, 256), 256, 0, stream>>>(h2, w3, b3, h3, B_, 512, 256, 1);
    bn_kernel<<<blk(256, 256), 256, 0, stream>>>(h3, g3, be3, B_, 256);
    head_kernel<<<1, 32, 0, stream>>>(h3, wo, bo, (float*)d_out, B_, 256);

    (void)in_sizes; (void)n_in; (void)out_size; (void)ws_size;
}
